// RGCNLayer_75677323755790
// MI455X (gfx1250) — matrix operand order repaired and checked by reading, not compiled, not run on hardware
//
#include <hip/hip_runtime.h>

// ---------------------------------------------------------------------------
// RGCN layer (basis decomposition) for MI455X / gfx1250, wave32 + WMMA.
//
// GEMM uses bf16x3 split-precision on V_WMMA_F32_16X16X32_BF16:
//   h = h_hi + h_lo (bf16 each), W = W_hi + W_lo (bf16 each)
//   h@W ~= h_hi@W_hi + h_hi@W_lo + h_lo@W_hi   (f32 accumulate)
// giving fp32-class accuracy at the bf16 matrix rate (8x K-depth per
// instruction vs the f32 WMMA path; 0.375x matrix-instruction count after
// the 3x term expansion).
//
// Pipeline:
//   0) zero d_out
//   1) split h -> h_hi/h_lo bf16
//   2) build weight from basis (+torch .view quirk), emit bf16 hi/lo
//      TRANSPOSED [r][o][i] so GEMM B-panels stage without LDS transposes
//   3) per relation r: t = h @ W[r]  (bf16x3 WMMA tiled GEMM)
//   4) per relation r: edges with etype==r: atomicAdd d_out[dst] += t[src]
//   5) d_out = relu(d_out + bias)
// ---------------------------------------------------------------------------

typedef __bf16 v16bf __attribute__((ext_vector_type(16)));
typedef float  v8f   __attribute__((ext_vector_type(8)));

#define IN_DIM    256
#define OUT_DIM   256
#define NN_NODES  20000
#define NN_EDGES  320000
#define NUM_RELS  8
#define NUM_BASES 4

#define KT   32            // K panel per LDS stage
#define PAD  40            // LDS row pitch in bf16 (80B: breaks bank conflicts,
                           //  keeps all fragment chunks 16B-aligned)

// --------------------------------------------------------------------- utils
__global__ void rgcn_zero_f32(float* __restrict__ p, int n) {
    int i = blockIdx.x * blockDim.x + threadIdx.x;
    if (i < n) p[i] = 0.0f;
}

// h -> bf16 hi/lo split (done once, reused by all 8 relation GEMMs)
__global__ void rgcn_split_h(const float* __restrict__ h,
                             __bf16* __restrict__ hh,
                             __bf16* __restrict__ hl, int n) {
    int i = blockIdx.x * blockDim.x + threadIdx.x;
    if (i >= n) return;
    float x  = h[i];
    __bf16 t = (__bf16)x;            // RNE
    hh[i] = t;
    hl[i] = (__bf16)(x - (float)t);
}

// weight[r][i][o] = sum_b w_comp[r][b] * basis_raw[(i*NUM_BASES+b)*OUT + o]
// (basis raw buffer [B,IN,OUT] reinterpreted as [IN,B,OUT], per reference).
// Output stored TRANSPOSED as wT[r][o][i], split into bf16 hi/lo.
__global__ void rgcn_build_weight(const float* __restrict__ basis,
                                  const float* __restrict__ w_comp,
                                  __bf16* __restrict__ wTh,
                                  __bf16* __restrict__ wTl) {
    int idx = blockIdx.x * blockDim.x + threadIdx.x;   // r*65536 + i*256 + o
    if (idx >= NUM_RELS * IN_DIM * OUT_DIM) return;
    int r = idx >> 16;
    int i = (idx >> 8) & 255;
    int o = idx & 255;
    float acc = 0.0f;
#pragma unroll
    for (int b = 0; b < NUM_BASES; ++b)
        acc += w_comp[r * NUM_BASES + b] * basis[(i * NUM_BASES + b) * OUT_DIM + o];
    __bf16 t = (__bf16)acc;
    int ti = r * (IN_DIM * OUT_DIM) + o * IN_DIM + i;  // transposed [r][o][i]
    wTh[ti] = t;
    wTl[ti] = (__bf16)(acc - (float)t);
}

// ------------------------------------------------------------------- GEMM
// t[n][o] = sum_d h[n][d] * W[d][o] for one relation, bf16x3.
// Block: 256 threads = 8 waves. Block tile: M=128, N=64. Wave tile: M=16,N=64
// (4 v8f accumulators). K staged through LDS in panels of KT=32 (one bf16
// WMMA K-depth).
//
// Fragment layouts used here (16-bit, 16x16x32; same k order for A and B):
//   A 16x32: lane(lo,hi): row M=lo; K range [16*hi..16*hi+15]
//            -> two contiguous 16B ds_load_b128 from row-major [m][k] panel
//   B 32x16: lane(lo,hi): col N=lo; K range [16*hi..16*hi+15]
//            -> one contiguous 32B (2x b128) from transposed [n][k] panel
//   C/D 16x16: VGPR v: lanes 0-15 -> (M=v, N=lane); 16-31 -> (M=v+8, N=lane-16)
__global__ __launch_bounds__(256)
void rgcn_gemm_wmma_bf16x3(const __bf16* __restrict__ hh,   // [NN][256]
                           const __bf16* __restrict__ hl,   // [NN][256]
                           const __bf16* __restrict__ wTh,  // [256 o][256 i]
                           const __bf16* __restrict__ wTl,  // [256 o][256 i]
                           float* __restrict__ t)           // [NN][256]
{
    __shared__ __align__(16) __bf16 HAh[128 * PAD];  // A hi: [m][k]
    __shared__ __align__(16) __bf16 HAl[128 * PAD];  // A lo
    __shared__ __align__(16) __bf16 WBh[64 * PAD];   // B hi: [n][k] (transposed)
    __shared__ __align__(16) __bf16 WBl[64 * PAD];   // B lo

    const int tid   = threadIdx.x;
    const int lane  = tid & 31;
    const int wave  = tid >> 5;
    const int lo    = lane & 15;
    const int hi    = lane >> 4;
    const int mBlk  = blockIdx.x * 128;
    const int nBlk  = blockIdx.y * 64;
    const int mWave = mBlk + wave * 16;

    v8f acc[4];
#pragma unroll
    for (int j = 0; j < 4; ++j)
#pragma unroll
        for (int v = 0; v < 8; ++v) acc[j][v] = 0.0f;

    for (int k0 = 0; k0 < IN_DIM; k0 += KT) {
        // ---- stage A panels: 128x32 bf16 hi+lo, 16B chunks, coalesced
#pragma unroll
        for (int p = 0; p < 2; ++p) {
            int idx = tid + p * 256;           // 0..511 chunks of 8 bf16
            int m = idx >> 2, k8 = (idx & 3) * 8;
            int row = mBlk + m;
            if (row >= NN_NODES) row = NN_NODES - 1;   // clamp tail (unused)
            *(uint4*)&HAh[m * PAD + k8] = *(const uint4*)&hh[row * IN_DIM + k0 + k8];
            *(uint4*)&HAl[m * PAD + k8] = *(const uint4*)&hl[row * IN_DIM + k0 + k8];
        }
        // ---- stage B panels: 64x32 bf16 hi+lo (already transposed in global)
        {
            int n = tid >> 2, k8 = (tid & 3) * 8;
            *(uint4*)&WBh[n * PAD + k8] = *(const uint4*)&wTh[(nBlk + n) * IN_DIM + k0 + k8];
            *(uint4*)&WBl[n * PAD + k8] = *(const uint4*)&wTl[(nBlk + n) * IN_DIM + k0 + k8];
        }
        // prefetch next A panel (gfx1250 global_prefetch_b8)
        if (k0 + KT < IN_DIM) {
            int row = mBlk + (tid >> 1);
            if (row >= NN_NODES) row = NN_NODES - 1;
            __builtin_prefetch(&hh[row * IN_DIM + k0 + KT + (tid & 1) * 16], 0, 1);
        }
        __syncthreads();

        // ---- A fragments (two 16B chunks each, k = 16*hi..16*hi+15, same order as B)
        v16bf ah, al;
        {
            const __bf16* rp = &HAh[(wave * 16 + lo) * PAD];
            ((uint4*)&ah)[0] = *(const uint4*)(rp + 16 * hi);
            ((uint4*)&ah)[1] = *(const uint4*)(rp + 16 * hi + 8);
            rp = &HAl[(wave * 16 + lo) * PAD];
            ((uint4*)&al)[0] = *(const uint4*)(rp + 16 * hi);
            ((uint4*)&al)[1] = *(const uint4*)(rp + 16 * hi + 8);
        }
#pragma unroll
        for (int j = 0; j < 4; ++j) {
            v16bf bh, bl;
            const __bf16* wp = &WBh[(j * 16 + lo) * PAD + 16 * hi];
            ((uint4*)&bh)[0] = *(const uint4*)(wp);
            ((uint4*)&bh)[1] = *(const uint4*)(wp + 8);
            wp = &WBl[(j * 16 + lo) * PAD + 16 * hi];
            ((uint4*)&bl)[0] = *(const uint4*)(wp);
            ((uint4*)&bl)[1] = *(const uint4*)(wp + 8);

            acc[j] = __builtin_amdgcn_wmma_f32_16x16x32_bf16(
                false, ah, false, bh, (short)0, acc[j], false, false);
            acc[j] = __builtin_amdgcn_wmma_f32_16x16x32_bf16(
                false, ah, false, bl, (short)0, acc[j], false, false);
            acc[j] = __builtin_amdgcn_wmma_f32_16x16x32_bf16(
                false, al, false, bh, (short)0, acc[j], false, false);
        }
        __syncthreads();
    }

    if (mWave < NN_NODES) {               // wave-uniform guard (EXEC stays full)
#pragma unroll
        for (int j = 0; j < 4; ++j) {
            int col = nBlk + j * 16 + lo;
#pragma unroll
            for (int v = 0; v < 8; ++v) {
                int row = mWave + v + 8 * hi;
                t[row * OUT_DIM + col] = acc[j][v];
            }
        }
    }
}

// --------------------------------------------------------------- scatter-sum
// For edges of relation `rel`: d_out[dst] += t[src].  64 threads per edge,
// float4 slice each; f32 global atomics (target stays L2-resident: 20.5MB).
__global__ __launch_bounds__(256)
void rgcn_scatter_rel(const float* __restrict__ t,
                      const int* __restrict__ src,
                      const int* __restrict__ dst,
                      const int* __restrict__ etype,
                      float* __restrict__ agg,
                      int rel)
{
    int e = blockIdx.x * 4 + (threadIdx.x >> 6);
    if (e >= NN_EDGES) return;
    if (etype[e] != rel) return;
    int sl = (threadIdx.x & 63) * 4;
    int s = src[e], d = dst[e];
    const float4 v = *(const float4*)&t[s * OUT_DIM + sl];
    float* o = &agg[d * OUT_DIM + sl];
    atomicAdd(o + 0, v.x);
    atomicAdd(o + 1, v.y);
    atomicAdd(o + 2, v.z);
    atomicAdd(o + 3, v.w);
}

// ------------------------------------------------------------------ finalize
__global__ void rgcn_finalize(float* __restrict__ out,
                              const float* __restrict__ bias, int n) {
    int i = blockIdx.x * blockDim.x + threadIdx.x;
    if (i >= n) return;
    float v = out[i] + bias[i & (OUT_DIM - 1)];
    out[i] = v > 0.0f ? v : 0.0f;
}

// ------------------------------------------------------------------- launch
extern "C" void kernel_launch(void* const* d_in, const int* in_sizes, int n_in,
                              void* d_out, int out_size, void* d_ws, size_t ws_size,
                              hipStream_t stream) {
    const float* h      = (const float*)d_in[0];   // [20000,256]
    const float* basis  = (const float*)d_in[1];   // [4,256,256]
    const float* w_comp = (const float*)d_in[2];   // [8,4]
    const float* bias   = (const float*)d_in[3];   // [256]
    const int*   src    = (const int*)d_in[4];     // [320000]
    const int*   dst    = (const int*)d_in[5];     // [320000]
    const int*   etype  = (const int*)d_in[6];     // [320000]
    float* out = (float*)d_out;                    // [20000,256]

    const int nH = NN_NODES * IN_DIM;      // 5,120,000
    const int nW = NUM_RELS * IN_DIM * OUT_DIM;  // 524,288

    // workspace: h_hi | h_lo | wT_hi | wT_lo | t   (~43 MB total)
    __bf16* hh  = (__bf16*)d_ws;
    __bf16* hl  = hh + nH;
    __bf16* wTh = hl + nH;
    __bf16* wTl = wTh + nW;
    float*  t   = (float*)(wTl + nW);

    const int nOut = NN_NODES * OUT_DIM;   // 5,120,000

    // 0) zero accumulator
    rgcn_zero_f32<<<(nOut + 255) / 256, 256, 0, stream>>>(out, nOut);

    // 1) split h into bf16 hi/lo
    rgcn_split_h<<<(nH + 255) / 256, 256, 0, stream>>>(h, hh, hl, nH);

    // 2) build transposed bf16 hi/lo weights
    rgcn_build_weight<<<(nW + 255) / 256, 256, 0, stream>>>(basis, w_comp, wTh, wTl);

    // 3/4) per relation: bf16x3 WMMA GEMM then edge scatter
    dim3 gemmGrid((NN_NODES + 127) / 128, OUT_DIM / 64);   // 157 x 4
    for (int r = 0; r < NUM_RELS; ++r) {
        rgcn_gemm_wmma_bf16x3<<<gemmGrid, 256, 0, stream>>>(
            hh, hl, wTh + r * IN_DIM * OUT_DIM, wTl + r * IN_DIM * OUT_DIM, t);
        rgcn_scatter_rel<<<NN_EDGES / 4, 256, 0, stream>>>(
            t, src, dst, etype, out, r);
    }

    // 5) bias + relu in place
    rgcn_finalize<<<(nOut + 255) / 256, 256, 0, stream>>>(out, bias, nOut);
}
